// local_policy_att_65111704207704
// MI455X (gfx1250) — compile-verified, matrix operand order checked by reading, not run on hardware
//
#include <hip/hip_runtime.h>
#include <hip/hip_bf16.h>
#include <math.h>

#define BDIM 256
#define Bsz 48
#define Wsz 100
#define Nsz 1000
#define EMB 128
#define Hh 8
#define QKVd 16
#define LOCALk 100
#define Lr 101   // depot + 100 neighbors
#define Lp 112   // Lr padded to multiple of 16
#define STRA 136 // half-stride for f16 A / Bt (16B aligned rows, bank-spread)
#define STRK 132 // float-stride for K/V outputs

typedef _Float16 v16h __attribute__((ext_vector_type(16)));
typedef _Float16 v8h  __attribute__((ext_vector_type(8)));
typedef float    v8f  __attribute__((ext_vector_type(8)));

// ln(10000)/63  (pos_enc inverse-frequency exponent step)
#define C_LOG 0.1461958789202569f
// 1/sqrt(128)
#define RS128 0.08838834764831843f

// Load one 16x32 f16 WMMA fragment slice for this lane.
// Layout (ISA 7.12.2, 16-bit A 16x32): lanes 0-15: K = {0..7, 16..23}+0;
// lanes 16-31: K = {0..7, 16..23}+8.  Two contiguous 8-half (16B) groups.
__device__ __forceinline__ v16h load_frag(const _Float16* rowp, int koff) {
  const v8h* p = (const v8h*)(rowp + koff);
  v8h lo = p[0];   // K = koff .. koff+7
  v8h hi = p[2];   // K = koff+16 .. koff+23
  v16h r;
#pragma unroll
  for (int z = 0; z < 8; ++z) { r[z] = lo[z]; r[z + 8] = hi[z]; }
  return r;
}

__device__ __forceinline__ bool kless(float av, int ai, float bv, int bi) {
  return (av < bv) || (av == bv && ai < bi);
}

__global__ __launch_bounds__(BDIM) void local_policy_att_kernel(
    const float* __restrict__ dist,
    const float* __restrict__ xy,
    const float* __restrict__ ndem,
    const float* __restrict__ nmask,
    const float* __restrict__ init_w,
    const float* __restrict__ init_b,
    const float* __restrict__ cur_token,
    const float* __restrict__ Wq,
    const float* __restrict__ Wk,
    const float* __restrict__ Wv,
    const float* __restrict__ Wc_w,
    const float* __restrict__ Wc_b,
    float* __restrict__ out)
{
  __shared__ float s_val[1024];
  __shared__ int   s_idx[1024];
  __shared__ float s_red[128];
  __shared__ float s_fx[Lp], s_fy[Lp], s_fd[Lp], s_sm[Lp];
  __shared__ int   s_gidx[Lp];
  __shared__ __attribute__((aligned(16))) _Float16 s_A[Lp * STRA];
  __shared__ __attribute__((aligned(16))) _Float16 s_Bt[EMB * STRA];
  __shared__ float s_K[Lp * STRK];
  __shared__ float s_V[Lp * STRK];
  __shared__ float s_q[EMB];
  __shared__ float s_att[Hh * Lp];
  __shared__ float s_ssum[Hh];
  __shared__ float s_out[EMB];
  __shared__ float s_mh[EMB];
  __shared__ float s_nfac;
  __shared__ int   s_non;

  const int t = threadIdx.x;
  const int bw = blockIdx.x;
  const size_t base = (size_t)bw * Nsz;
  float* orow = out + base;

  // zero this block's output row (scatter happens after many barriers)
  for (int i = t; i < Nsz; i += BDIM) orow[i] = 0.0f;

  // ---- Phase A: dist_eff for nodes 1..N-1, pad to 1024, bitonic sort ----
  for (int i = t; i < 1024; i += BDIM) {
    float v;
    if (i < Nsz - 1) v = dist[base + i + 1] - nmask[base + i + 1];
    else             v = __builtin_inff();     // pad: +inf, index >= 999
    s_val[i] = v;
    s_idx[i] = i;
  }
  __syncthreads();
  for (int k = 2; k <= 1024; k <<= 1) {
    for (int j = k >> 1; j > 0; j >>= 1) {
      for (int i = t; i < 1024; i += BDIM) {
        int ixj = i ^ j;
        if (ixj > i) {
          float av = s_val[i], bv = s_val[ixj];
          int   ai = s_idx[i], bi = s_idx[ixj];
          bool dirAsc = ((i & k) == 0);
          bool dosw = dirAsc ? kless(bv, bi, av, ai) : kless(av, ai, bv, bi);
          if (dosw) {
            s_val[i] = bv;  s_idx[i] = bi;
            s_val[ixj] = av; s_idx[ixj] = ai;
          }
        }
      }
      __syncthreads();
    }
  }

  // ---- Phase B: norm factor from 100 smallest (inf -> 0) ----
  {
    float m = -__builtin_inff();
    if (t < LOCALk) {
      float v = s_val[t];
      if (isinf(v)) v = 0.0f;
      m = v;
    }
    if (t < 128) s_red[t] = m;
    __syncthreads();
    for (int s = 64; s > 0; s >>= 1) {
      if (t < s) s_red[t] = fmaxf(s_red[t], s_red[t + s]);
      __syncthreads();
    }
    if (t == 0) { s_nfac = s_red[0] + 1e-6f; s_non = (s_red[0] != 0.0f) ? 1 : 0; }
  }
  __syncthreads();

  // ---- Phase C: gather features; compute shared query q = cur_token @ Wq ----
  if (t < Lp) {
    float fx = 0.0f, fy = 0.0f, fd = 0.0f, sm = 0.0f;
    int gi = 0;
    if (t < Lr) {
      float sd;
      if (t == 0) { gi = 0; sd = __builtin_inff(); }       // depot always masked
      else        { gi = s_idx[t - 1] + 1; sd = s_val[t - 1]; }
      sm = nmask[base + gi];
      if (!isinf(sd)) {
        fd = ndem[base + gi];
        fx = xy[2 * (base + gi)];
        fy = xy[2 * (base + gi) + 1];
        if (s_non) { fx /= s_nfac; fy /= s_nfac; }
      }
    }
    s_fx[t] = fx; s_fy[t] = fy; s_fd[t] = fd; s_sm[t] = sm; s_gidx[t] = gi;
  }
  if (t < EMB) {
    float acc = 0.0f;
    for (int c = 0; c < EMB; ++c) acc += cur_token[c] * Wq[c * EMB + t];
    s_q[t] = acc;
  }
  __syncthreads();

  // ---- Phase D: init_k = feat@init_w + b + pos_enc  ->  f16 A matrix ----
  for (int p = t; p < Lp * EMB; p += BDIM) {
    int l = p >> 7, e = p & 127;
    float v = 0.0f;
    if (l < Lr) {
      int i = e & 63;
      float freq = expf(-(float)i * C_LOG);
      float ang = (float)l * freq;
      float pe = (e < 64) ? sinf(ang) : cosf(ang);
      v = s_fx[l] * init_w[e] + s_fy[l] * init_w[EMB + e]
        + s_fd[l] * init_w[2 * EMB + e] + init_b[e] + pe;
    }
    s_A[l * STRA + e] = (_Float16)v;
  }
  __syncthreads();

  // ---- Phase E: K = A@Wk, V = A@Wv via v_wmma_f32_16x16x32_f16 ----
  const int wid = t >> 5, lane = t & 31;
  const int nn = lane & 15;
  const int kb = (lane < 16) ? 0 : 8;     // per-lane K base (ISA A/B layout)
  for (int pass = 0; pass < 2; ++pass) {
    const float* Wsrc = pass ? Wv : Wk;
    for (int p = t; p < EMB * EMB; p += BDIM) {   // coalesced read, transposed store
      int k = p >> 7, n = p & 127;
      s_Bt[n * STRA + k] = (_Float16)Wsrc[p];
    }
    __syncthreads();
    float* dst = pass ? s_V : s_K;
    const _Float16* brow = &s_Bt[(wid * 16 + nn) * STRA];  // wave owns N-tile = wid
    v16h bf0 = load_frag(brow, 0 * 32 + kb);
    v16h bf1 = load_frag(brow, 1 * 32 + kb);
    v16h bf2 = load_frag(brow, 2 * 32 + kb);
    v16h bf3 = load_frag(brow, 3 * 32 + kb);
    for (int mt = 0; mt < 7; ++mt) {
      v8f acc = {};
      const _Float16* arow = &s_A[(mt * 16 + nn) * STRA];
      v16h a0 = load_frag(arow, 0 * 32 + kb);
      acc = __builtin_amdgcn_wmma_f32_16x16x32_f16(false, a0, false, bf0, (short)0, acc, false, false);
      v16h a1 = load_frag(arow, 1 * 32 + kb);
      acc = __builtin_amdgcn_wmma_f32_16x16x32_f16(false, a1, false, bf1, (short)0, acc, false, false);
      v16h a2 = load_frag(arow, 2 * 32 + kb);
      acc = __builtin_amdgcn_wmma_f32_16x16x32_f16(false, a2, false, bf2, (short)0, acc, false, false);
      v16h a3 = load_frag(arow, 3 * 32 + kb);
      acc = __builtin_amdgcn_wmma_f32_16x16x32_f16(false, a3, false, bf3, (short)0, acc, false, false);
      // C/D layout: VGPR r -> rows M=r (lanes 0-15) / M=8+r (lanes 16-31)
      int mrow = mt * 16 + ((lane >= 16) ? 8 : 0);
      int col  = wid * 16 + nn;
#pragma unroll
      for (int r = 0; r < 8; ++r)
        dst[(mrow + r) * STRK + col] = acc[r];
    }
    __syncthreads();
  }

  // ---- Phase F: attention scores + softmax (wave h owns head h) ----
  {
    const int h = wid;
    float lmax = -__builtin_inff();
    float sc[4];
#pragma unroll
    for (int z = 0; z < 4; ++z) {
      int l = lane + 32 * z;
      float s = -__builtin_inff();
      if (l < Lr) {
        float acc = 0.0f;
        const float* kp = &s_K[l * STRK + h * QKVd];
#pragma unroll
        for (int d = 0; d < QKVd; ++d) acc += s_q[h * QKVd + d] * kp[d];
        s = acc * 0.25f + s_sm[l];        // 1/sqrt(16)
      }
      sc[z] = s;
      lmax = fmaxf(lmax, s);
    }
#pragma unroll
    for (int off = 16; off > 0; off >>= 1)
      lmax = fmaxf(lmax, __shfl_xor(lmax, off, 32));
    float lsum = 0.0f;
#pragma unroll
    for (int z = 0; z < 4; ++z) {
      int l = lane + 32 * z;
      if (l < Lr) {
        float e = expf(sc[z] - lmax);
        s_att[h * Lp + l] = e;
        lsum += e;
      }
    }
#pragma unroll
    for (int off = 16; off > 0; off >>= 1)
      lsum += __shfl_xor(lsum, off, 32);
    if (lane == 0) s_ssum[h] = lsum;
  }
  __syncthreads();

  // ---- Phase G: out = softmax @ V ----
  if (t < EMB) {
    int h = t >> 4;
    float acc = 0.0f;
    for (int l = 0; l < Lr; ++l) acc += s_att[h * Lp + l] * s_V[l * STRK + t];
    s_out[t] = acc / s_ssum[h];
  }
  __syncthreads();

  // ---- Phase H: mh = out @ Wc_w + Wc_b ----
  if (t < EMB) {
    float acc = Wc_b[t];
    for (int c = 0; c < EMB; ++c) acc += s_out[c] * Wc_w[c * EMB + t];
    s_mh[t] = acc;
  }
  __syncthreads();

  // ---- Phase I: score2 = mh . init_k / sqrt(128), scatter to idx ----
  if (t < Lr) {
    const int l = t;
    float acc = 0.0f;
    for (int e = 0; e < EMB; ++e) {
      int i = e & 63;
      float freq = expf(-(float)i * C_LOG);
      float ang = (float)l * freq;
      float pe = (e < 64) ? sinf(ang) : cosf(ang);
      float ik = s_fx[l] * init_w[e] + s_fy[l] * init_w[EMB + e]
               + s_fd[l] * init_w[2 * EMB + e] + init_b[e] + pe;
      acc += s_mh[e] * ik;
    }
    orow[s_gidx[l]] = acc * RS128;
  }
}

extern "C" void kernel_launch(void* const* d_in, const int* in_sizes, int n_in,
                              void* d_out, int out_size, void* d_ws, size_t ws_size,
                              hipStream_t stream) {
  (void)in_sizes; (void)n_in; (void)out_size; (void)d_ws; (void)ws_size;
  const float* dist      = (const float*)d_in[1];
  const float* xy        = (const float*)d_in[2];
  const float* ndem      = (const float*)d_in[3];
  const float* nmask     = (const float*)d_in[4];
  const float* init_w    = (const float*)d_in[5];
  const float* init_b    = (const float*)d_in[6];
  const float* cur_token = (const float*)d_in[7];
  const float* Wq        = (const float*)d_in[8];
  const float* Wk        = (const float*)d_in[9];
  const float* Wv        = (const float*)d_in[10];
  const float* Wc_w      = (const float*)d_in[11];
  const float* Wc_b      = (const float*)d_in[12];
  local_policy_att_kernel<<<Bsz * Wsz, BDIM, 0, stream>>>(
      dist, xy, ndem, nmask, init_w, init_b, cur_token,
      Wq, Wk, Wv, Wc_w, Wc_b, (float*)d_out);
}
